// SenriAttention_39084202394229
// MI455X (gfx1250) — matrix operand order repaired and checked by reading, not compiled, not run on hardware
//
#include <hip/hip_runtime.h>
#include <math.h>

typedef _Float16 h16;
typedef __attribute__((ext_vector_type(16))) _Float16 v16h;
typedef __attribute__((ext_vector_type(8)))  _Float16 h8;
typedef __attribute__((ext_vector_type(8)))  float    v8f;

#define S_LEN  2048
#define HIDDEN 2048
#define NH     16
#define NKV    4
#define HD     128
#define NQKV   3072   // H*D + 2*KV*D
#define WIN    1024

// Per-lane async copy of 16 bytes global -> LDS (ASYNCcnt-tracked, no VGPR data).
// Generic addresses of __shared__ objects carry the LDS offset in the low 32 bits
// (ISA 10.2: LDS address is the low 32 address bits), so truncation yields a valid VDST value.
__device__ __forceinline__ void async_g2l_b128(void* lds_dst, const void* gsrc) {
  unsigned loff = (unsigned)(size_t)lds_dst;
  asm volatile("global_load_async_to_lds_b128 %0, %1, off"
               :: "v"(loff), "v"(gsrc) : "memory");
}

__device__ __forceinline__ void wait_async0() {
#if __has_builtin(__builtin_amdgcn_s_wait_asynccnt)
  __builtin_amdgcn_s_wait_asynccnt(0);
#else
  asm volatile("s_wait_asynccnt 0x0" ::: "memory");
#endif
}

// ---------------------------------------------------------------- conversions
__global__ void k_f32_to_f16(const float* __restrict__ src, h16* __restrict__ dst, int n) {
  int i = (blockIdx.x * blockDim.x + threadIdx.x) * 4;
  if (i + 3 < n) {
    float4 v = *(const float4*)(src + i);
    dst[i + 0] = (h16)v.x;  dst[i + 1] = (h16)v.y;
    dst[i + 2] = (h16)v.z;  dst[i + 3] = (h16)v.w;
  }
}

__global__ void k_pack_qkv_w(const float* __restrict__ Wq, const float* __restrict__ Wk,
                             const float* __restrict__ Wv, h16* __restrict__ out) {
  int idx = blockIdx.x * blockDim.x + threadIdx.x;
  if (idx >= HIDDEN * NQKV) return;
  int k = idx / NQKV, n = idx % NQKV;
  float v;
  if (n < 2048)      v = Wq[(size_t)k * 2048 + n];
  else if (n < 2560) v = Wk[(size_t)k * 512 + (n - 2048)];
  else               v = Wv[(size_t)k * 512 + (n - 2560)];
  out[idx] = (h16)v;
}

__global__ void k_pack_bias(const float* __restrict__ bq, const float* __restrict__ bk,
                            const float* __restrict__ bv, float* __restrict__ out) {
  int n = blockIdx.x * blockDim.x + threadIdx.x;
  if (n >= NQKV) return;
  float v;
  if (n < 2048)      v = bq[n];
  else if (n < 2560) v = bk[n - 2048];
  else               v = bv[n - 2560];
  out[n] = v;
}

// ---------------------------------------------------------------- WMMA GEMM
// C[M,N] = A[M,K] * B[K,N] (+bias).  Block tile 128x128, 256 thr = 8 waves
// (wave grid 4(M strips of 32) x 2(N strips of 64)); K step = 32 per WMMA.
// Double-buffered LDS: next tile's A goes via async DMA and B via registers
// while the current tile's WMMAs execute; drained with s_wait_asynccnt.
// M and N must be multiples of 128 (true for all call sites).
template <bool OUT_F32, bool HAS_BIAS>
__global__ __launch_bounds__(256) void k_gemm_f16(
    const h16* __restrict__ A, const h16* __restrict__ B,
    const float* __restrict__ bias, h16* __restrict__ Ch,
    float* __restrict__ Cf, int M, int N, int K) {
  __shared__ h16 lds_a[2][128 * 32];   // [m][k]
  __shared__ h16 lds_b[2][128 * 32];   // transposed: [n][k]
  const int tid = threadIdx.x;
  const int lane = tid & 31, wave = tid >> 5;
  const int lane16 = lane & 15, hi16 = lane >> 4;   // wave32 halves
  const int m0 = blockIdx.y * 128, n0 = blockIdx.x * 128;
  const int wm = wave & 3, wn = wave >> 2;

  // fixed per-thread staging coordinates
  const int aRow[2] = { (tid) >> 2, (tid + 256) >> 2 };
  const int aC8[2]  = { ((tid) & 3) * 8, ((tid + 256) & 3) * 8 };
  const int bR[2]   = { (tid) >> 4, (tid + 256) >> 4 };
  const int bC8[2]  = { ((tid) & 15) * 8, ((tid + 256) & 15) * 8 };

  v8f acc[2][4];
#pragma unroll
  for (int a = 0; a < 2; ++a)
#pragma unroll
    for (int b = 0; b < 4; ++b)
#pragma unroll
      for (int r = 0; r < 8; ++r) acc[a][b][r] = 0.f;

  auto stage_a = [&](int k0, int buf) {
#pragma unroll
    for (int i = 0; i < 2; ++i) {
      const h16* gp = A + (size_t)(m0 + aRow[i]) * K + k0 + aC8[i];
      async_g2l_b128(&lds_a[buf][aRow[i] * 32 + aC8[i]], gp);
    }
  };
  auto load_b = [&](int k0, h8* breg) {
#pragma unroll
    for (int i = 0; i < 2; ++i)
      breg[i] = *(const h8*)(B + (size_t)(k0 + bR[i]) * N + n0 + bC8[i]);
  };
  auto store_b = [&](const h8* breg, int buf) {
#pragma unroll
    for (int i = 0; i < 2; ++i)
#pragma unroll
      for (int t = 0; t < 8; ++t)
        lds_b[buf][(bC8[i] + t) * 32 + bR[i]] = breg[i][t];
  };
  auto compute = [&](int buf) {
    v16h af[2], bf[4];
#pragma unroll
    for (int mi = 0; mi < 2; ++mi) {
      const h16* p = &lds_a[buf][(wm * 32 + mi * 16 + lane16) * 32];
      h8 lo = *(const h8*)(p + hi16 * 8);
      h8 hv = *(const h8*)(p + 16 + hi16 * 8);
#pragma unroll
      for (int t = 0; t < 8; ++t) { af[mi][t] = lo[t]; af[mi][8 + t] = hv[t]; }
    }
#pragma unroll
    for (int ni = 0; ni < 4; ++ni) {
      const h16* p = &lds_b[buf][(wn * 64 + ni * 16 + lane16) * 32 + hi16 * 8];
      h8 lo = *(const h8*)p;
      h8 hv = *(const h8*)(p + 16);
#pragma unroll
      for (int t = 0; t < 8; ++t) { bf[ni][t] = lo[t]; bf[ni][8 + t] = hv[t]; }
    }
#pragma unroll
    for (int mi = 0; mi < 2; ++mi)
#pragma unroll
      for (int ni = 0; ni < 4; ++ni)
        acc[mi][ni] = __builtin_amdgcn_wmma_f32_16x16x32_f16(
            false, af[mi], false, bf[ni], (short)0, acc[mi][ni], false, false);
  };

  // prologue: stage tile 0 into buffer 0
  {
    stage_a(0, 0);
    h8 breg[2];
    load_b(0, breg);
    store_b(breg, 0);
    wait_async0();
    __syncthreads();
  }

  int buf = 0;
  for (int k0 = 0; k0 + 32 < K; k0 += 32) {
    // stage next tile while current computes
    stage_a(k0 + 32, buf ^ 1);
    h8 breg[2];
    load_b(k0 + 32, breg);
    compute(buf);
    store_b(breg, buf ^ 1);
    wait_async0();
    __syncthreads();
    buf ^= 1;
  }
  compute(buf);   // last tile

#pragma unroll
  for (int mi = 0; mi < 2; ++mi)
#pragma unroll
    for (int ni = 0; ni < 4; ++ni)
#pragma unroll
      for (int r = 0; r < 8; ++r) {
        int gm = m0 + wm * 32 + mi * 16 + r + hi16 * 8;
        int gn = n0 + wn * 64 + ni * 16 + lane16;
        float v = acc[mi][ni][r];
        if constexpr (HAS_BIAS) v += bias[gn];
        if constexpr (OUT_F32) Cf[(size_t)gm * N + gn] = v;
        else                   Ch[(size_t)gm * N + gn] = (h16)v;
      }
}

// ---------------------------------------------------------------- RoPE / V
__global__ void k_rope_q(const h16* __restrict__ qkv, h16* __restrict__ q_rot) {
  int idx = blockIdx.x * blockDim.x + threadIdx.x;
  if (idx >= S_LEN * NH * 64) return;
  int j = idx & 63, h = (idx >> 6) & 15, s = idx >> 10;
  float inv = __expf(-0.14391157f * (float)j);   // 10000^(-2j/128)
  float ang = (float)s * inv;
  float c = __cosf(ang), sn = __sinf(ang);
  float x1 = (float)qkv[(size_t)s * NQKV + h * HD + j];
  float x2 = (float)qkv[(size_t)s * NQKV + h * HD + j + 64];
  size_t o = ((size_t)h * S_LEN + s) * HD + j;
  q_rot[o]      = (h16)(x1 * c - x2 * sn);
  q_rot[o + 64] = (h16)(x2 * c + x1 * sn);
}

__global__ void k_rope_k(const h16* __restrict__ qkv, h16* __restrict__ k_rot) {
  int idx = blockIdx.x * blockDim.x + threadIdx.x;
  if (idx >= S_LEN * NKV * 64) return;
  int j = idx & 63, kv = (idx >> 6) & 3, s = idx >> 8;
  float inv = __expf(-0.14391157f * (float)j);
  float ang = (float)s * inv;
  float c = __cosf(ang), sn = __sinf(ang);
  float x1 = (float)qkv[(size_t)s * NQKV + 2048 + kv * HD + j];
  float x2 = (float)qkv[(size_t)s * NQKV + 2048 + kv * HD + j + 64];
  size_t o = ((size_t)kv * S_LEN + s) * HD + j;
  k_rot[o]      = (h16)(x1 * c - x2 * sn);
  k_rot[o + 64] = (h16)(x2 * c + x1 * sn);
}

__global__ void k_copy_v(const h16* __restrict__ qkv, h16* __restrict__ v_h) {
  int idx = blockIdx.x * blockDim.x + threadIdx.x;
  if (idx >= S_LEN * NKV * HD) return;
  int d = idx & 127, kv = (idx >> 7) & 3, s = idx >> 9;
  v_h[((size_t)kv * S_LEN + s) * HD + d] = qkv[(size_t)s * NQKV + 2560 + kv * HD + d];
}

// ---------------------------------------------------------------- flash attn
// grid (S/64, H), 128 thr = 4 waves; each wave owns 16 query rows.
__global__ __launch_bounds__(128) void k_flash(
    const h16* __restrict__ q_rot, const h16* __restrict__ k_rot,
    const h16* __restrict__ v_h, const float* __restrict__ mgate,
    h16* __restrict__ attn_out) {
  __shared__ h16 lds_k[64 * 128];    // [key][d]
  __shared__ h16 lds_vt[128 * 64];   // [d][key]
  __shared__ h16 lds_p[4 * 16 * 64]; // per-wave P tiles
  const int tid = threadIdx.x;
  const int lane = tid & 31, wave = tid >> 5;
  const int lane16 = lane & 15, hi16 = lane >> 4;
  const int qt = blockIdx.x, h = blockIdx.y, kvh = h >> 2;
  const int q0 = qt * 64 + wave * 16;
  const float SCL = 0.08838834764831845f;   // 128^-0.5

  // Q fragments: 16 rows x 128 d as 4 chunks of K=32
  v16h qf[4];
  {
    const h16* qp = q_rot + ((size_t)h * S_LEN + q0 + lane16) * HD;
#pragma unroll
    for (int c = 0; c < 4; ++c) {
      h8 lo = *(const h8*)(qp + 32 * c + hi16 * 8);
      h8 hv = *(const h8*)(qp + 32 * c + 16 + hi16 * 8);
#pragma unroll
      for (int t = 0; t < 8; ++t) { qf[c][t] = lo[t]; qf[c][8 + t] = hv[t]; }
    }
  }

  v8f o[8];
  float m[8], l[8], corr[8];
#pragma unroll
  for (int n = 0; n < 8; ++n)
#pragma unroll
    for (int r = 0; r < 8; ++r) o[n][r] = 0.f;
#pragma unroll
  for (int r = 0; r < 8; ++r) { m[r] = -1e30f; l[r] = 0.f; }

  int lo_key = qt * 64 - (WIN - 1); if (lo_key < 0) lo_key = 0;
  int kb_lo = (lo_key / 64) * 64;

  for (int kb = kb_lo; kb < qt * 64 + 64; kb += 64) {
    // stage K rows kb..kb+127, d 0..63 (row stride 128) via async global->LDS
#pragma unroll
    for (int i = 0; i < 8; ++i) {
      int cid = tid + 128 * i;
      int row = cid >> 3, c8 = (cid & 7) * 8;
      async_g2l_b128(lds_k + row * 128 + c8,
                     k_rot + ((size_t)kvh * S_LEN + kb + row) * HD + c8);
    }
    // stage V transposed [d][key] (needs per-element transpose)
#pragma unroll
    for (int i = 0; i < 8; ++i) {
      int e8 = tid + 128 * i;
      int row = e8 >> 4, d8 = (e8 & 15) * 8;
      h8 v = *(const h8*)(v_h + ((size_t)kvh * S_LEN + kb + row) * HD + d8);
#pragma unroll
      for (int t = 0; t < 8; ++t) lds_vt[(d8 + t) * 64 + row] = v[t];
    }
    wait_async0();
    __syncthreads();

    // S = Q K^T  (16 rows x 64 keys per wave)
    v8f sacc[4];
#pragma unroll
    for (int n = 0; n < 4; ++n)
#pragma unroll
      for (int r = 0; r < 8; ++r) sacc[n][r] = 0.f;
#pragma unroll
    for (int c = 0; c < 4; ++c)
#pragma unroll
      for (int n = 0; n < 4; ++n) {
        v16h bf;
        const h16* p = lds_k + (16 * n + lane16) * 128 + 32 * c + hi16 * 8;
        h8 lo = *(const h8*)p;
        h8 hv = *(const h8*)(p + 16);
#pragma unroll
        for (int t = 0; t < 8; ++t) { bf[t] = lo[t]; bf[8 + t] = hv[t]; }
        sacc[n] = __builtin_amdgcn_wmma_f32_16x16x32_f16(
            false, qf[c], false, bf, (short)0, sacc[n], false, false);
      }

    // scale + sliding-window causal mask
#pragma unroll
    for (int n = 0; n < 4; ++n)
#pragma unroll
      for (int r = 0; r < 8; ++r) {
        int qi = q0 + r + hi16 * 8;
        int kj = kb + 16 * n + lane16;
        float s = sacc[n][r] * SCL;
        bool ok = (kj <= qi) && (qi - kj < WIN);
        sacc[n][r] = ok ? s : -1e30f;
      }

    // online softmax (row reductions across the 16-lane half)
#pragma unroll
    for (int r = 0; r < 8; ++r) {
      float mx = fmaxf(fmaxf(sacc[0][r], sacc[1][r]), fmaxf(sacc[2][r], sacc[3][r]));
#pragma unroll
      for (int msk = 1; msk < 16; msk <<= 1) mx = fmaxf(mx, __shfl_xor(mx, msk, 32));
      float mn = fmaxf(m[r], mx);
      corr[r] = __expf(m[r] - mn);
      m[r] = mn;
    }
#pragma unroll
    for (int r = 0; r < 8; ++r) {
      float ls = 0.f;
#pragma unroll
      for (int n = 0; n < 4; ++n) {
        float sv = sacc[n][r];
        float p = (sv <= -1e29f) ? 0.f : __expf(sv - m[r]);
        ls += p;
        lds_p[(wave * 16 + r + hi16 * 8) * 64 + 16 * n + lane16] = (h16)p;
      }
#pragma unroll
      for (int msk = 1; msk < 16; msk <<= 1) ls += __shfl_xor(ls, msk, 32);
      l[r] = l[r] * corr[r] + ls;
    }
#pragma unroll
    for (int n = 0; n < 8; ++n)
#pragma unroll
      for (int r = 0; r < 8; ++r) o[n][r] *= corr[r];

    // O += P V  (contraction over 64 keys, 2 chunks of 32)
#pragma unroll
    for (int kc = 0; kc < 2; ++kc) {
      v16h pf;
      const h16* pp = lds_p + (wave * 16 + lane16) * 64 + 32 * kc;
      h8 plo = *(const h8*)(pp + hi16 * 8);
      h8 phi = *(const h8*)(pp + 16 + hi16 * 8);
#pragma unroll
      for (int t = 0; t < 8; ++t) { pf[t] = plo[t]; pf[8 + t] = phi[t]; }
#pragma unroll
      for (int n = 0; n < 8; ++n) {
        v16h vf;
        const h16* vp = lds_vt + (16 * n + lane16) * 64 + 32 * kc + hi16 * 8;
        h8 vlo = *(const h8*)vp;
        h8 vhi = *(const h8*)(vp + 16);
#pragma unroll
        for (int t = 0; t < 8; ++t) { vf[t] = vlo[t]; vf[8 + t] = vhi[t]; }
        o[n] = __builtin_amdgcn_wmma_f32_16x16x32_f16(
            false, pf, false, vf, (short)0, o[n], false, false);
      }
    }
    __syncthreads();
  }

  // epilogue: out = (1 - sigmoid(gate[h])) * local   (global path is exactly 0)
  float g = mgate[h];
  float wloc = 1.f - 1.f / (1.f + __expf(-g));
#pragma unroll
  for (int r = 0; r < 8; ++r) {
    float inv = wloc / l[r];
#pragma unroll
    for (int n = 0; n < 8; ++n) {
      size_t off = (size_t)(q0 + r + hi16 * 8) * HIDDEN + h * HD + 16 * n + lane16;
      attn_out[off] = (h16)(o[n][r] * inv);
    }
  }
}

// ---------------------------------------------------------------- launch
extern "C" void kernel_launch(void* const* d_in, const int* in_sizes, int n_in,
                              void* d_out, int out_size, void* d_ws, size_t ws_size,
                              hipStream_t stream) {
  (void)in_sizes; (void)n_in; (void)out_size; (void)ws_size;
  const float* hs = (const float*)d_in[0];
  const float* Wq = (const float*)d_in[1];
  const float* bq = (const float*)d_in[2];
  const float* Wk = (const float*)d_in[3];
  const float* bk = (const float*)d_in[4];
  const float* Wv = (const float*)d_in[5];
  const float* bv = (const float*)d_in[6];
  const float* Wo = (const float*)d_in[7];
  const float* mg = (const float*)d_in[8];
  float* out = (float*)d_out;

  char* ws = (char*)d_ws;
  size_t off = 0;
  auto carve = [&](size_t bytes) -> void* {
    void* p = ws + off;
    off = (off + bytes + 255) & ~(size_t)255;
    return p;
  };
  h16*   hs_h   = (h16*)  carve((size_t)S_LEN * HIDDEN * 2);
  h16*   wqkv_h = (h16*)  carve((size_t)HIDDEN * NQKV * 2);
  float* bqkv   = (float*)carve((size_t)NQKV * 4);
  h16*   qkv_h  = (h16*)  carve((size_t)S_LEN * NQKV * 2);
  h16*   q_rot  = (h16*)  carve((size_t)NH * S_LEN * HD * 2);
  h16*   k_rot  = (h16*)  carve((size_t)NKV * S_LEN * HD * 2);
  h16*   v_h    = (h16*)  carve((size_t)NKV * S_LEN * HD * 2);
  h16*   attn_h = (h16*)  carve((size_t)S_LEN * HIDDEN * 2);
  h16*   wo_h   = (h16*)  carve((size_t)HIDDEN * HIDDEN * 2);

  int n1 = S_LEN * HIDDEN;
  k_f32_to_f16<<<(n1 / 4 + 255) / 256, 256, 0, stream>>>(hs, hs_h, n1);
  int n2 = HIDDEN * HIDDEN;
  k_f32_to_f16<<<(n2 / 4 + 255) / 256, 256, 0, stream>>>(Wo, wo_h, n2);
  int n3 = HIDDEN * NQKV;
  k_pack_qkv_w<<<(n3 + 255) / 256, 256, 0, stream>>>(Wq, Wk, Wv, wqkv_h);
  k_pack_bias<<<(NQKV + 255) / 256, 256, 0, stream>>>(bq, bk, bv, bqkv);

  dim3 g1(NQKV / 128, S_LEN / 128);
  k_gemm_f16<false, true><<<g1, 256, 0, stream>>>(hs_h, wqkv_h, bqkv, qkv_h, nullptr,
                                                  S_LEN, NQKV, HIDDEN);

  int nq = S_LEN * NH * 64;
  k_rope_q<<<(nq + 255) / 256, 256, 0, stream>>>(qkv_h, q_rot);
  int nk = S_LEN * NKV * 64;
  k_rope_k<<<(nk + 255) / 256, 256, 0, stream>>>(qkv_h, k_rot);
  int nv = S_LEN * NKV * HD;
  k_copy_v<<<(nv + 255) / 256, 256, 0, stream>>>(qkv_h, v_h);

  dim3 g2(S_LEN / 64, NH);
  k_flash<<<g2, 128, 0, stream>>>(q_rot, k_rot, v_h, mg, attn_h);

  dim3 g3(HIDDEN / 128, S_LEN / 128);
  k_gemm_f16<true, false><<<g3, 256, 0, stream>>>(attn_h, wo_h, nullptr, nullptr, out,
                                                  S_LEN, HIDDEN, HIDDEN);
}
